// MHA_13477607375200
// MI455X (gfx1250) — matrix operand order repaired and checked by reading, not compiled, not run on hardware
//
#include <hip/hip_runtime.h>

// ---------------------------------------------------------------------------
// MHA forward for gfx1250 (MI455X, CDNA5).
//  - all matmuls: v_wmma_f32_16x16x32_bf16 (f32 accum)
//  - weights/activations pre-converted to bf16; B always [n][k] (K-contig)
//  - LDS staging via GLOBAL_LOAD_ASYNC_TO_LDS_B128 (ASYNCcnt) where the
//    source is bf16; f32->bf16 convert path only where f32 is mandatory (attn)
//  - double-buffered LDS, one barrier per K-step
//  - f32->bf16 via v_cvt_pk_bf16_f32 when the builtin is available
// ---------------------------------------------------------------------------

typedef __attribute__((ext_vector_type(16))) __bf16           v16bf;
typedef __attribute__((ext_vector_type(8)))  float            v8f;
typedef __attribute__((ext_vector_type(8)))  unsigned short   v8us;

#define MHA_B   2
#define MHA_TQ  2048
#define MHA_TK  2048
#define MHA_D   1024
#define MHA_H   16
#define MHA_DH  64

#define BM 128
#define BK 32
#define THREADS 256
#define LDS_STRIDE 40   // 32 + 8 pad (ushorts); rows are 80B => 16B aligned

// fp32 -> bf16 bits, round-to-nearest-even (scalar fallback)
__device__ __forceinline__ unsigned int f2bf_bits(float f) {
  union { float f; unsigned u; } x; x.f = f;
  return (x.u + 0x7FFFu + ((x.u >> 16) & 1u)) >> 16;
}
// two floats -> packed 2xbf16 (one v_cvt_pk_bf16_f32 when available)
__device__ __forceinline__ unsigned pack2bf(float a, float b) {
#if __has_builtin(__builtin_amdgcn_cvt_pk_bf16_f32)
  typedef __attribute__((ext_vector_type(2))) __bf16 v2bf;
  union { v2bf v; unsigned u; } r;
  r.v = __builtin_amdgcn_cvt_pk_bf16_f32(a, b);
  return r.u;
#else
  return f2bf_bits(a) | (f2bf_bits(b) << 16);
#endif
}
__device__ __forceinline__ uint2 pack4bf(float4 f) {
  uint2 r;
  r.x = pack2bf(f.x, f.y);
  r.y = pack2bf(f.z, f.w);
  return r;
}

// async 16B copy global(bf16) -> LDS, tracked on ASYNCcnt
__device__ __forceinline__ void async_cp16(unsigned ldsByteOff, const unsigned short* g) {
  asm volatile("global_load_async_to_lds_b128 %0, %1, off"
               :: "v"(ldsByteOff), "v"(g) : "memory");
}
__device__ __forceinline__ void wait_async() {
  asm volatile("s_wait_asynccnt 0x0" ::: "memory");
}

// A fragment: 16x32 bf16. lane 0-15: row=lane, K halves {0..7,16..23};
// lane 16-31: row=lane-16, K halves {8..15,24..31}.   (ISA 7.12.2)
__device__ __forceinline__ v16bf load_frag_a(const unsigned short* As, int rowBase, int lane) {
  int row = rowBase + (lane & 15);
  int kh  = (lane >> 4) << 3;
  union { v16bf v; v8us h[2]; } u;
  u.h[0] = *(const v8us*)(As + row * LDS_STRIDE + kh);
  u.h[1] = *(const v8us*)(As + row * LDS_STRIDE + 16 + kh);
  return u.v;
}
// B fragment: 32x16 bf16, LDS [n][k] K-contiguous.
// lane 0-15: col=lane, K {0..7,16..23}; lane 16-31: col=lane-16, K {8..15,24..31}.
__device__ __forceinline__ v16bf load_frag_b(const unsigned short* Bs, int colBase, int lane) {
  int col = colBase + (lane & 15);
  int ko  = (lane >> 4) << 3;
  union { v16bf v; v8us h[2]; } u;
  u.h[0] = *(const v8us*)(Bs + col * LDS_STRIDE + ko);
  u.h[1] = *(const v8us*)(Bs + col * LDS_STRIDE + ko + 16);
  return u.v;
}

// ---------------------------------------------------------------------------
// Generic batched GEMM:  C[bh] = alpha * A[bh] * Bt[bh]^T (+ bias[n])
//  A: MxK row-major (f32 if !ABF, bf16 if ABF), lda
//  Bt: NxK row-major bf16 (K-contiguous), ldb
//  C: f32 MxN (ldc) if !OUTBF; bf16 if OUTBF; TSTORE => C[gc*ldc + gr]
// Per-z: base + b*s?b + h*s?h (element strides), b=z/H, h=z%H.
// ---------------------------------------------------------------------------
template <int BN_T, bool ABF, bool OUTBF, bool TSTORE>
__global__ void __launch_bounds__(THREADS)
gemm_wmma(const void* __restrict__ Ab, const unsigned short* __restrict__ Btb,
          const float* __restrict__ bias, void* __restrict__ Cb,
          int lda, int ldb, int ldc, int kTotal, float alpha, int H,
          long long sAb, long long sAh, long long sBb, long long sBh,
          long long sCb, long long sCh) {
  constexpr int WN      = BN_T / 2;            // wave tile N (2 waves along N)
  constexpr int FRAG_N  = WN / 16;             // 4 (BN=128) or 2 (BN=64)
  constexpr int ASB     = BM * LDS_STRIDE * 2; // bytes per A buffer (10240)
  constexpr int BSB     = BN_T * LDS_STRIDE * 2;
  constexpr int BS_BASE = 2 * ASB;
  constexpr int BCH     = (BN_T * 4) / THREADS;  // 16B chunks/thread for B

  __shared__ __align__(16) unsigned char smem[2 * ASB + 2 * BSB];

  const int tid  = threadIdx.x;
  const int lane = tid & 31;
  const int wid  = tid >> 5;
  const int waveM = wid & 3;
  const int waveN = wid >> 2;

  const int z = blockIdx.z;
  const int b = z / H;
  const int h = z % H;

  const float*          Af32 = nullptr;
  const unsigned short* Ab16 = nullptr;
  if constexpr (ABF) Ab16 = (const unsigned short*)Ab + b * sAb + h * sAh;
  else               Af32 = (const float*)Ab + b * sAb + h * sAh;
  const unsigned short* Bt = Btb + b * sBb + h * sBh;
  float*          Cf  = nullptr;
  unsigned short* Cus = nullptr;
  if constexpr (OUTBF) Cus = (unsigned short*)Cb + b * sCb + h * sCh;
  else                 Cf  = (float*)Cb + b * sCb + h * sCh;

  const int mBase = blockIdx.y * BM;
  const int nBase = blockIdx.x * BN_T;

  v8f acc[2][FRAG_N] = {};

  auto stage = [&](int buf, int k0) {
    // ---- A tile -----------------------------------------------------------
    if constexpr (ABF) {                       // async bf16 copy, 512 chunks
#pragma unroll
      for (int i = 0; i < 2; ++i) {
        int c = i * THREADS + tid;
        int r = c >> 2, seg = (c & 3) << 3;    // 8 bf16 per 16B chunk
        async_cp16((unsigned)(buf * ASB + r * (LDS_STRIDE * 2) + seg * 2),
                   Ab16 + (long long)(mBase + r) * lda + (k0 + seg));
      }
    } else {                                   // f32 -> bf16 convert path
      unsigned short* As = (unsigned short*)(smem + buf * ASB);
#pragma unroll
      for (int i = 0; i < 4; ++i) {
        int idx = i * THREADS + tid;
        int r = idx >> 3, k4 = (idx & 7) << 2;
        float4 f = *(const float4*)(Af32 + (long long)(mBase + r) * lda + (k0 + k4));
        *(uint2*)(&As[r * LDS_STRIDE + k4]) = pack4bf(f);
      }
    }
    // ---- B tile (always bf16 [n][k], async) -------------------------------
#pragma unroll
    for (int i = 0; i < BCH; ++i) {
      int c = i * THREADS + tid;
      int n = c >> 2, seg = (c & 3) << 3;
      async_cp16((unsigned)(BS_BASE + buf * BSB + n * (LDS_STRIDE * 2) + seg * 2),
                 Bt + (long long)(nBase + n) * ldb + (k0 + seg));
    }
  };

  // ---- main loop: ping-pong, 1 barrier / K-step ----------------------------
  stage(0, 0);
  wait_async();
  __syncthreads();
  int buf = 0;
  for (int k0 = 0; k0 < kTotal; k0 += BK) {
    if (k0 + BK < kTotal) stage(buf ^ 1, k0 + BK);

    const unsigned short* As = (const unsigned short*)(smem + buf * ASB);
    const unsigned short* Bs = (const unsigned short*)(smem + BS_BASE + buf * BSB);
    v16bf aF0 = load_frag_a(As, waveM * 32, lane);
    v16bf aF1 = load_frag_a(As, waveM * 32 + 16, lane);
    v16bf bF[FRAG_N];
#pragma unroll
    for (int j = 0; j < FRAG_N; ++j)
      bF[j] = load_frag_b(Bs, waveN * WN + j * 16, lane);

#pragma unroll
    for (int ni = 0; ni < FRAG_N; ++ni) {
      acc[0][ni] = __builtin_amdgcn_wmma_f32_16x16x32_bf16(false, aF0, false, bF[ni], (short)0, acc[0][ni], false, false);
      acc[1][ni] = __builtin_amdgcn_wmma_f32_16x16x32_bf16(false, aF1, false, bF[ni], (short)0, acc[1][ni], false, false);
    }
    wait_async();          // next buffer's async copies complete
    __syncthreads();       // all waves done reading buf / writing buf^1
    buf ^= 1;
  }

  // ---- epilogue. D layout: VGPR r -> row (lane<16 ? r : r+8), col=lane&15 --
  const int rShift = (lane >> 4) * 8;
  const int colIn  = lane & 15;
#pragma unroll
  for (int mi = 0; mi < 2; ++mi) {
#pragma unroll
    for (int ni = 0; ni < FRAG_N; ++ni) {
      int gr = mBase + waveM * 32 + mi * 16 + rShift;
      int gc = nBase + waveN * WN + ni * 16 + colIn;
      float bv = bias ? bias[gc] : 0.0f;
#pragma unroll
      for (int r = 0; r < 8; ++r) {
        float val = acc[mi][ni][r] * alpha + bv;
        long long off = TSTORE ? (long long)gc * ldc + (gr + r)
                               : (long long)(gr + r) * ldc + gc;
        if constexpr (OUTBF) Cus[off] = (unsigned short)(pack2bf(val, 0.0f) & 0xFFFFu);
        else                 Cf[off]  = val;
      }
    }
  }
}

// ---------------------------------------------------------------------------
// Weight transpose + f32->bf16:  dst[n][k] = bf16(src[k][n]),  dim x dim
// ---------------------------------------------------------------------------
__global__ void __launch_bounds__(256)
transpose_cvt_kernel(const float* __restrict__ src, unsigned short* __restrict__ dst, int dim) {
  __shared__ float t[32][33];
  const int bx = blockIdx.x * 32, by = blockIdx.y * 32;
  const int tx = threadIdx.x & 31, ty = threadIdx.x >> 5;   // 32 x 8
#pragma unroll
  for (int j = 0; j < 32; j += 8)
    t[ty + j][tx] = src[(long long)(by + ty + j) * dim + (bx + tx)];
  __syncthreads();
#pragma unroll
  for (int j = 0; j < 32; j += 8)
    dst[(long long)(bx + ty + j) * dim + (by + tx)] =
        (unsigned short)(pack2bf(t[tx][ty + j], 0.0f) & 0xFFFFu);
}

// ---------------------------------------------------------------------------
// In-place masked softmax over attn rows. One 8-wave block per row.
// ---------------------------------------------------------------------------
__device__ __forceinline__ float waveMax(float v) {
#pragma unroll
  for (int o = 16; o > 0; o >>= 1) v = fmaxf(v, __shfl_xor(v, o, 32));
  return v;
}
__device__ __forceinline__ float waveSum(float v) {
#pragma unroll
  for (int o = 16; o > 0; o >>= 1) v += __shfl_xor(v, o, 32);
  return v;
}

__global__ void __launch_bounds__(256)
softmax_mask_kernel(float* __restrict__ attn, const int* __restrict__ mask) {
  __shared__ float red[8];
  const long long row = blockIdx.x;
  float* p = attn + row * (long long)MHA_TK;
  const int b = (int)(row / ((long long)MHA_H * MHA_TQ));
  const int q = (int)(row % MHA_TQ);
  const int* mrow = mask + ((long long)b * MHA_TQ + q) * (long long)MHA_TK;

  const int tid = threadIdx.x, lane = tid & 31, wid = tid >> 5;
  const float NEG_INF = -__builtin_inff();

  float lmax = NEG_INF;
#pragma unroll
  for (int i = 0; i < MHA_TK / 256; ++i) {
    int c = i * 256 + tid;
    float v = mrow[c] ? p[c] : NEG_INF;
    lmax = fmaxf(lmax, v);
  }
  float wm = waveMax(lmax);
  if (lane == 0) red[wid] = wm;
  __syncthreads();
  float rmax;
  { float t = (lane < 8) ? red[lane] : NEG_INF; rmax = waveMax(t); }
  __syncthreads();

  float lsum = 0.0f;
#pragma unroll
  for (int i = 0; i < MHA_TK / 256; ++i) {
    int c = i * 256 + tid;
    float v = mrow[c] ? p[c] : NEG_INF;
    float e = __expf(v - rmax);
    p[c] = e;
    lsum += e;
  }
  float ws = waveSum(lsum);
  if (lane == 0) red[wid] = ws;
  __syncthreads();
  float rsum;
  { float t = (lane < 8) ? red[lane] : 0.0f; rsum = waveSum(t); }
  float inv = 1.0f / rsum;
#pragma unroll
  for (int i = 0; i < MHA_TK / 256; ++i) {
    int c = i * 256 + tid;
    p[c] *= inv;
  }
}

// ---------------------------------------------------------------------------
extern "C" void kernel_launch(void* const* d_in, const int* in_sizes, int n_in,
                              void* d_out, int out_size, void* d_ws, size_t ws_size,
                              hipStream_t stream) {
  const float* q    = (const float*)d_in[0];
  const float* k    = (const float*)d_in[1];
  const float* v    = (const float*)d_in[2];
  const int*   mask = (const int*)d_in[3];
  const float* w_q  = (const float*)d_in[4];
  const float* b_q  = (const float*)d_in[5];
  const float* w_k  = (const float*)d_in[6];
  const float* b_k  = (const float*)d_in[7];
  const float* w_v  = (const float*)d_in[8];
  const float* b_v  = (const float*)d_in[9];
  const float* w_o  = (const float*)d_in[10];
  const float* b_o  = (const float*)d_in[11];

  const long long actE = (long long)MHA_B * MHA_TQ * MHA_D;   // 4,194,304
  const long long wE   = (long long)MHA_D * MHA_D;            // 1,048,576
  float* out  = (float*)d_out;                                // [B,TQ,D]
  float* attn = out + actE;                                   // [B,H,TQ,TK] f32

  unsigned short* wqT = (unsigned short*)d_ws;    // [D][D]  (n-major, k-contig)
  unsigned short* wkT = wqT + wE;
  unsigned short* wvT = wkT + wE;
  unsigned short* woT = wvT + wE;
  unsigned short* Qp  = woT + wE;                 // [B*TQ][D] bf16
  unsigned short* Kp  = Qp + actE;                // [B*TK][D] bf16
  unsigned short* Vt  = Kp + actE;                // [B][D][TK] bf16 (transposed)
  unsigned short* Op  = Vt + actE;                // [B*TQ][D] bf16

  const long long sAttnB = (long long)MHA_H * MHA_TQ * MHA_TK;
  const long long sAttnH = (long long)MHA_TQ * MHA_TK;
  const long long sActB  = (long long)MHA_TK * MHA_D;
  const float scale = 0.125f;                     // 64^-0.5
  dim3 blk(THREADS);

  // 0) weight f32 -> bf16 transposed (one-time, K-contiguous for async B path)
  dim3 gT(MHA_D / 32, MHA_D / 32);
  transpose_cvt_kernel<<<gT, blk, 0, stream>>>(w_q, wqT, MHA_D);
  transpose_cvt_kernel<<<gT, blk, 0, stream>>>(w_k, wkT, MHA_D);
  transpose_cvt_kernel<<<gT, blk, 0, stream>>>(w_v, wvT, MHA_D);
  transpose_cvt_kernel<<<gT, blk, 0, stream>>>(w_o, woT, MHA_D);

  // 1-2) Q/K projections -> bf16 activations
  dim3 gProj(MHA_D / 128, (MHA_B * MHA_TQ) / BM, 1);
  gemm_wmma<128, false, true, false><<<gProj, blk, 0, stream>>>(q, wqT, b_q, Qp,
      MHA_D, MHA_D, MHA_D, MHA_D, 1.0f, 1, 0, 0, 0, 0, 0, 0);
  gemm_wmma<128, false, true, false><<<gProj, blk, 0, stream>>>(k, wkT, b_k, Kp,
      MHA_D, MHA_D, MHA_D, MHA_D, 1.0f, 1, 0, 0, 0, 0, 0, 0);

  // 3) V projection, stored transposed: Vt[b][d][t]
  dim3 gProjV(MHA_D / 128, MHA_TQ / BM, MHA_B);
  gemm_wmma<128, false, true, true><<<gProjV, blk, 0, stream>>>(v, wvT, b_v, Vt,
      MHA_D, MHA_D, /*ldc(T)=*/MHA_TK, MHA_D, 1.0f, 1,
      (long long)MHA_TQ * MHA_D, 0, 0, 0, (long long)MHA_D * MHA_TK, 0);

  // 4) scores = scale * Qh * Kh^T  -> d_out attn region (f32)
  dim3 gScore(MHA_TK / 128, MHA_TQ / BM, MHA_B * MHA_H);
  gemm_wmma<128, true, false, false><<<gScore, blk, 0, stream>>>(Qp, Kp, nullptr, attn,
      MHA_D, MHA_D, MHA_TK, MHA_DH, scale, MHA_H,
      sActB, MHA_DH, sActB, MHA_DH, sAttnB, sAttnH);

  // 5) masked softmax in place
  dim3 gSm(MHA_B * MHA_H * MHA_TQ);
  softmax_mask_kernel<<<gSm, dim3(256), 0, stream>>>(attn, mask);

  // 6) Oh = attn * Vh  (A = f32 attn, B = Vt async) -> bf16 Op
  dim3 gAV(MHA_DH / 64, MHA_TQ / BM, MHA_B * MHA_H);
  gemm_wmma<64, false, true, false><<<gAV, blk, 0, stream>>>(attn, Vt, nullptr, Op,
      MHA_TK, MHA_TK, MHA_D, MHA_TK, 1.0f, MHA_H,
      sAttnB, sAttnH, (long long)MHA_D * MHA_TK, (long long)MHA_DH * MHA_TK,
      sActB, MHA_DH);

  // 7) out = Op * w_o + b_o  (all-bf16 async inputs, f32 out)
  gemm_wmma<128, true, false, false><<<gProj, blk, 0, stream>>>(Op, woT, b_o, out,
      MHA_D, MHA_D, MHA_D, MHA_D, 1.0f, 1, 0, 0, 0, 0, 0, 0);
}
